// MessagePassing_43387759624641
// MI455X (gfx1250) — matrix operand order repaired and checked by reading, not compiled, not run on hardware
//
#include <hip/hip_runtime.h>

typedef __attribute__((ext_vector_type(16))) _Float16 v16h;
typedef __attribute__((ext_vector_type(8)))  float    v8f;

constexpr int N_NODES = 50000;
constexpr int N_EDGES = 800000;
constexpr int D    = 128;
constexpr int RDIM = 32;
constexpr int AC   = 192;   // msg mlp width  (msg_src|radial|angular)
constexpr int EC   = 320;   // edge mlp width (src|dst|radial|angular)

union AFrag { v16h v; _Float16 h[16]; };
union BFrag { v16h v; uint4 q[2]; };

// sigmoid via v_exp_f32 + v_rcp_f32 (avoid the IEEE div_scale/div_fixup chain)
static __device__ __forceinline__ float sigmoidf(float x) {
    return __builtin_amdgcn_rcpf(1.0f + __expf(-x));
}

// Build one 16x32 f16 A-fragment for this lane from 32 consecutive f32.
// A order used here (contiguous k, same as the B staging): lanes 0-15 -> M=0..15; elems 0..15 = K(half*16+j).
static __device__ __forceinline__ void load_a_frag(AFrag& a, const float* base, int half) {
    const float4* p0 = reinterpret_cast<const float4*>(base + half * 16);
    const float4* p1 = reinterpret_cast<const float4*>(base + half * 16 + 8);
    float4 f0 = p0[0], f1 = p0[1], f2 = p1[0], f3 = p1[1];
    a.h[0]  = (_Float16)f0.x; a.h[1]  = (_Float16)f0.y;
    a.h[2]  = (_Float16)f0.z; a.h[3]  = (_Float16)f0.w;
    a.h[4]  = (_Float16)f1.x; a.h[5]  = (_Float16)f1.y;
    a.h[6]  = (_Float16)f1.z; a.h[7]  = (_Float16)f1.w;
    a.h[8]  = (_Float16)f2.x; a.h[9]  = (_Float16)f2.y;
    a.h[10] = (_Float16)f2.z; a.h[11] = (_Float16)f2.w;
    a.h[12] = (_Float16)f3.x; a.h[13] = (_Float16)f3.y;
    a.h[14] = (_Float16)f3.z; a.h[15] = (_Float16)f3.w;
}

// Stage weight [K x N] (row-major f32) into LDS as f16 in exact B-fragment order:
// frag(kc,nt) base = (kc*NT+nt)*512 halfs; lane L holds 16 contiguous halfs: K=(L>>4)*16+e, N=nt*16+(L&15).
template <int KDIM, int NDIM>
static __device__ __forceinline__ void stage_weights(const float* __restrict__ w, _Float16* wlds) {
    constexpr int NT = NDIM / 16;
    for (int idx = threadIdx.x; idx < KDIM * NDIM; idx += 256) {
        int k  = idx / NDIM, n = idx - k * NDIM;
        int kc = k >> 5, kr = k & 31, nt = n >> 4, nc = n & 15;
        int lane = ((kr >> 4) << 4) | nc;
        int e    = kr & 15;
        wlds[((kc * NT + nt) << 9) + (lane << 4) + e] = (_Float16)w[idx];
    }
}

// ---------------- Kernel 1: edge message = sigmoid([msg_src|rad|ang] @ w1 + b1), scatter-add to messages[dst]
__global__ __launch_bounds__(256, 1)
void msg_kernel(const float* __restrict__ nf, const float* __restrict__ er,
                const float* __restrict__ ea, const int* __restrict__ ei,
                const float* __restrict__ w1, const float* __restrict__ b1,
                float* __restrict__ msgs) {
    constexpr int KC = AC / 32;   // 6
    constexpr int NT = AC / 16;   // 12
    __shared__ _Float16 wlds[KC * NT * 512];   // 73.7 KB
    __shared__ float    blds[AC];

    stage_weights<AC, AC>(w1, wlds);
    for (int i = threadIdx.x; i < AC; i += 256) blds[i] = b1[i];
    __syncthreads();

    const int lane = threadIdx.x & 31;
    const int half = lane >> 4;
    const int row  = lane & 15;                 // A-row (M) and D-col (N) for this lane
    const int tile = blockIdx.x * 8 + (threadIdx.x >> 5);
    const int e0   = tile * 16;

    const int eL = e0 + row;
    const int s1 = ei[eL];                      // src[e]
    const int s2 = ei[s1];                      // src[src[e]]  (reference quirk)

    AFrag a[KC];
#pragma unroll
    for (int kc = 0; kc < KC; ++kc) {
        const float* base;
        if (kc < 4)       base = nf + (size_t)s2 * D + kc * 32;
        else if (kc == 4) base = er + (size_t)eL * RDIM;
        else              base = ea + (size_t)eL * RDIM;
        load_a_frag(a[kc], base, half);
    }

    int drow[8];
#pragma unroll
    for (int j = 0; j < 8; ++j) drow[j] = ei[N_EDGES + e0 + half * 8 + j];

    for (int nt = 0; nt < NT; ++nt) {
        const float bv = blds[nt * 16 + row];
        v8f acc = {bv, bv, bv, bv, bv, bv, bv, bv};
#pragma unroll
        for (int kc = 0; kc < KC; ++kc) {
            BFrag b;
            const uint4* bp = reinterpret_cast<const uint4*>(&wlds[((kc * NT + nt) << 9) + (lane << 4)]);
            b.q[0] = bp[0]; b.q[1] = bp[1];
            acc = __builtin_amdgcn_wmma_f32_16x16x32_f16(false, a[kc].v, false, b.v, (short)0, acc, false, false);
        }
#pragma unroll
        for (int j = 0; j < 8; ++j) {
            float s = sigmoidf(acc[j]);
            atomicAdd(&msgs[(size_t)drow[j] * AC + nt * 16 + row], s);
        }
    }
}

// ---------------- Kernel 2: per-node mlp2 chain + self/neigh update (scalar; ~3% of FLOPs)
__global__ __launch_bounds__(64)
void node_kernel(const float* __restrict__ nf, const float* __restrict__ msgs,
                 const float* __restrict__ w2a, const float* __restrict__ b2a,
                 const float* __restrict__ w2b, const float* __restrict__ b2b,
                 const float* __restrict__ w2c, const float* __restrict__ b2c,
                 const float* __restrict__ w2d, const float* __restrict__ b2d,
                 const float* __restrict__ wself, const float* __restrict__ bself,
                 const float* __restrict__ wneigh, const float* __restrict__ bneigh,
                 float* __restrict__ out_node) {
    __shared__ float sb[528 * 64];   // thread-private columns: [feature][tid] -> conflict-free
    const int tid = threadIdx.x;
    const int n = blockIdx.x * 64 + tid;
    if (n >= N_NODES) return;
#define S(off, k) sb[((off) + (k)) * 64 + tid]
    constexpr int OA = 0, OB = 192, OC = 304, ODd = 336, OM = 400, ONF = 0;

    for (int k = 0; k < 192; ++k) S(OA, k) = msgs[(size_t)n * 192 + k];

    for (int j = 0; j < 112; ++j) {                     // 192 -> 112, relu
        float acc = b2a[j];
        for (int k = 0; k < 192; ++k) acc += S(OA, k) * w2a[k * 112 + j];
        S(OB, j) = fmaxf(acc, 0.0f);
    }
    for (int j = 0; j < 32; ++j) {                      // 112 -> 32, relu
        float acc = b2b[j];
        for (int k = 0; k < 112; ++k) acc += S(OB, k) * w2b[k * 32 + j];
        S(OC, j) = fmaxf(acc, 0.0f);
    }
    for (int j = 0; j < 64; ++j) {                      // 32 -> 64, relu
        float acc = b2c[j];
        for (int k = 0; k < 32; ++k) acc += S(OC, k) * w2c[k * 64 + j];
        S(ODd, j) = fmaxf(acc, 0.0f);
    }
    for (int j = 0; j < 128; ++j) {                     // 64 -> 128
        float acc = b2d[j];
        for (int k = 0; k < 64; ++k) acc += S(ODd, k) * w2d[k * 128 + j];
        S(OM, j) = acc;
    }
    for (int k = 0; k < 128; ++k) S(ONF, k) = nf[(size_t)n * 128 + k];   // OA region dead now

    for (int j = 0; j < 128; ++j) {
        float acc = bself[j] + bneigh[j];
        for (int k = 0; k < 128; ++k)
            acc += S(ONF, k) * wself[k * 128 + j] + S(OM, k) * wneigh[k * 128 + j];
        out_node[(size_t)n * 128 + j] = sigmoidf(acc) + S(ONF, j);
    }
#undef S
}

// ---------------- Kernel 3: edge_upd = sigmoid([src|dst|rad|ang] @ wedge + bedge) — dominant GEMM
__global__ __launch_bounds__(256, 1)
void edge_kernel(const float* __restrict__ nf, const float* __restrict__ er,
                 const float* __restrict__ ea, const int* __restrict__ ei,
                 const float* __restrict__ we, const float* __restrict__ be,
                 float* __restrict__ eout) {
    constexpr int KC = EC / 32;   // 10
    constexpr int NT = EC / 16;   // 20
    __shared__ _Float16 wlds[KC * NT * 512];   // 204.8 KB (<= 320 KB/WGP on CDNA5)
    __shared__ float    blds[EC];

    stage_weights<EC, EC>(we, wlds);
    for (int i = threadIdx.x; i < EC; i += 256) blds[i] = be[i];
    __syncthreads();

    const int lane = threadIdx.x & 31;
    const int half = lane >> 4;
    const int row  = lane & 15;
    const int tile = blockIdx.x * 8 + (threadIdx.x >> 5);
    const int e0   = tile * 16;

    const int eL = e0 + row;
    const int s1 = ei[eL];
    const int d1 = ei[N_EDGES + eL];

    AFrag a[KC];
#pragma unroll
    for (int kc = 0; kc < KC; ++kc) {
        const float* base;
        if (kc < 4)       base = nf + (size_t)s1 * D + kc * 32;
        else if (kc < 8)  base = nf + (size_t)d1 * D + (kc - 4) * 32;
        else if (kc == 8) base = er + (size_t)eL * RDIM;
        else              base = ea + (size_t)eL * RDIM;
        load_a_frag(a[kc], base, half);
    }

    for (int nt = 0; nt < NT; ++nt) {
        const float bv = blds[nt * 16 + row];
        v8f acc = {bv, bv, bv, bv, bv, bv, bv, bv};
#pragma unroll
        for (int kc = 0; kc < KC; ++kc) {
            BFrag b;
            const uint4* bp = reinterpret_cast<const uint4*>(&wlds[((kc * NT + nt) << 9) + (lane << 4)]);
            b.q[0] = bp[0]; b.q[1] = bp[1];
            acc = __builtin_amdgcn_wmma_f32_16x16x32_f16(false, a[kc].v, false, b.v, (short)0, acc, false, false);
        }
#pragma unroll
        for (int j = 0; j < 8; ++j) {
            float s = sigmoidf(acc[j]);
            eout[(size_t)(e0 + half * 8 + j) * EC + nt * 16 + row] = s;
        }
    }
}

extern "C" void kernel_launch(void* const* d_in, const int* in_sizes, int n_in,
                              void* d_out, int out_size, void* d_ws, size_t ws_size,
                              hipStream_t stream) {
    const float* nf     = (const float*)d_in[0];
    const float* er     = (const float*)d_in[1];
    const float* ea     = (const float*)d_in[2];
    const int*   ei     = (const int*)d_in[3];
    const float* w1     = (const float*)d_in[4];
    const float* b1     = (const float*)d_in[5];
    const float* w2a    = (const float*)d_in[6];
    const float* b2a    = (const float*)d_in[7];
    const float* w2b    = (const float*)d_in[8];
    const float* b2b    = (const float*)d_in[9];
    const float* w2c    = (const float*)d_in[10];
    const float* b2c    = (const float*)d_in[11];
    const float* w2d    = (const float*)d_in[12];
    const float* b2d    = (const float*)d_in[13];
    const float* wself  = (const float*)d_in[14];
    const float* bself  = (const float*)d_in[15];
    const float* wneigh = (const float*)d_in[16];
    const float* bneigh = (const float*)d_in[17];
    const float* wedge  = (const float*)d_in[18];
    const float* bedge  = (const float*)d_in[19];

    float* out_node = (float*)d_out;
    float* out_edge = out_node + (size_t)N_NODES * D;
    float* msgs     = (float*)d_ws;

    hipMemsetAsync(msgs, 0, (size_t)N_NODES * AC * sizeof(float), stream);

    const int edge_blocks = N_EDGES / (16 * 8);   // 6250, exact
    msg_kernel<<<edge_blocks, 256, 0, stream>>>(nf, er, ea, ei, w1, b1, msgs);
    node_kernel<<<(N_NODES + 63) / 64, 64, 0, stream>>>(nf, msgs, w2a, b2a, w2b, b2b,
                                                        w2c, b2c, w2d, b2d,
                                                        wself, bself, wneigh, bneigh, out_node);
    edge_kernel<<<edge_blocks, 256, 0, stream>>>(nf, er, ea, ei, wedge, bedge, out_edge);
}
